// SynthesisLayer_75247827026740
// MI455X (gfx1250) — compile-verified, matrix operand order checked by reading, not run on hardware
//
#include <hip/hip_runtime.h>

typedef _Float16 half_t;
typedef __attribute__((ext_vector_type(16))) _Float16 v16h;
typedef __attribute__((ext_vector_type(8)))  _Float16 v8h;
typedef __attribute__((ext_vector_type(8)))  float    v8f;
typedef __attribute__((ext_vector_type(4)))  float    v4f;

#define NN    16
#define HH    64
#define WW    64
#define CIN   512
#define COUT  512
#define WDIM  512
#define NUM_WS 18
#define LAYER_IDX 8
#define AFF_SCALE 0.04419417382415922f      /* 1/sqrt(512)   */
#define W_SCALE   0.014731391274719739f     /* 1/sqrt(9*512) */
#define LRELU_GAIN 1.4142135623730951f
#define CLIPV 256.0f

union AFrag { unsigned int u[8]; v16h h; };

// ---------------------------------------------------------------------------
// s[n][ci] = dlatents[n, LAYER_IDX] . (affine_w/sqrt(512)) + affine_b
// ---------------------------------------------------------------------------
__global__ void affine_kernel(const float* __restrict__ dlat,
                              const float* __restrict__ aw,
                              const float* __restrict__ ab,
                              float* __restrict__ s) {
    int tid = blockIdx.x * blockDim.x + threadIdx.x;   // 16*512
    int n  = tid >> 9;
    int ci = tid & 511;
    const float* z = dlat + (n * NUM_WS + LAYER_IDX) * WDIM;
    float acc = 0.f;
    for (int k = 0; k < WDIM; ++k) acc += z[k] * aw[k * CIN + ci];
    s[tid] = acc * AFF_SCALE + ab[ci];
}

// ---------------------------------------------------------------------------
// wsq[ci][co] = sum_tap (conv_w * W_SCALE)^2
// ---------------------------------------------------------------------------
__global__ void wsq_kernel(const float* __restrict__ cw,
                           float* __restrict__ wsq) {
    int tid = blockIdx.x * blockDim.x + threadIdx.x;   // 512*512
    float acc = 0.f;
    for (int t = 0; t < 9; ++t) {
        float v = cw[t * (CIN * COUT) + tid] * W_SCALE;
        acc += v * v;
    }
    wsq[tid] = acc;
}

// ---------------------------------------------------------------------------
// d[n][co] = rsqrt( sum_ci s[n][ci]^2 * wsq[ci][co] + 1e-8 )
// ---------------------------------------------------------------------------
__global__ void demod_kernel(const float* __restrict__ s,
                             const float* __restrict__ wsq,
                             float* __restrict__ d) {
    int tid = blockIdx.x * blockDim.x + threadIdx.x;   // 16*512
    int n  = tid >> 9;
    int co = tid & 511;
    float acc = 0.f;
    for (int ci = 0; ci < CIN; ++ci) {
        float sv = s[n * CIN + ci];
        acc += sv * sv * wsq[ci * COUT + co];
    }
    d[tid] = rsqrtf(acc + 1e-8f);
}

// ---------------------------------------------------------------------------
// wht[tap][co][ci] = (half)(conv_w[tap][ci][co] * W_SCALE)   (transposed for
// contiguous async-to-LDS B tiles)
// ---------------------------------------------------------------------------
__global__ void w2h_kernel(const float* __restrict__ cw,
                           half_t* __restrict__ wht) {
    int tid = blockIdx.x * blockDim.x + threadIdx.x;   // 9*512*512
    int tap = tid >> 18;
    int r   = tid & ((CIN * COUT) - 1);
    int co  = r >> 9;
    int ci  = r & 511;
    wht[tid] = (half_t)(cw[(tap * CIN + ci) * COUT + co] * W_SCALE);
}

// ---------------------------------------------------------------------------
// xm[n,h,w,ci] = (half)(x * s[n,ci])   (67 MB: L2-resident on MI455X)
// ---------------------------------------------------------------------------
__global__ void xmod_kernel(const float* __restrict__ x,
                            const float* __restrict__ s,
                            half_t* __restrict__ xm) {
    size_t base = ((size_t)blockIdx.x * blockDim.x + threadIdx.x) * 8;
    int ci  = (int)(base & 511);
    int pix = (int)(base >> 9);
    int n   = pix >> 12;                   // 64*64 pixels per sample
    v4f x0 = *(const v4f*)(x + base);
    v4f x1 = *(const v4f*)(x + base + 4);
    v4f s0 = *(const v4f*)(s + n * CIN + ci);
    v4f s1 = *(const v4f*)(s + n * CIN + ci + 4);
    v8h p;
    p[0] = (half_t)(x0.x * s0.x); p[1] = (half_t)(x0.y * s0.y);
    p[2] = (half_t)(x0.z * s0.z); p[3] = (half_t)(x0.w * s0.w);
    p[4] = (half_t)(x1.x * s1.x); p[5] = (half_t)(x1.y * s1.y);
    p[6] = (half_t)(x1.z * s1.z); p[7] = (half_t)(x1.w * s1.w);
    *(v8h*)(xm + base) = p;
}

__global__ void zero_kernel(unsigned int* __restrict__ z) {
    z[threadIdx.x] = 0u;                   // 256 * 4B = 512 halves
}

// ---------------------------------------------------------------------------
// Async implicit-GEMM 3x3 conv: block = row(n,h) x 128 couts, wave = 32x32.
// A and B tiles both staged with GLOBAL_LOAD_ASYNC_TO_LDS_B128; inner loop is
// async issues + ds_load_b32 fragments + 8 chained v_wmma_f32_16x16x32_f16.
// ---------------------------------------------------------------------------
__global__ void __launch_bounds__(256)
modconv_async_kernel(const half_t* __restrict__ xm,
                     const half_t* __restrict__ wht,
                     const half_t* __restrict__ zbuf,
                     const float*  __restrict__ dbuf,
                     const float*  __restrict__ cbias,
                     const float*  __restrict__ noise,
                     const float*  __restrict__ nstr,
                     float*        __restrict__ out) {
    __shared__ __align__(16) half_t Asmem[64 * 72];   // [pixel][ci 0..63]
    __shared__ __align__(16) half_t Bsmem[128 * 72];  // [cout ][ci 0..63]

    const int bid   = blockIdx.x;            // 1024 * 4
    const int tileN = bid & 3;
    const int tileM = bid >> 2;
    const int n   = tileM >> 6;
    const int h   = tileM & 63;
    const int co0 = tileN * 128;

    const int lane  = threadIdx.x & 31;
    const int wave  = threadIdx.x >> 5;
    const int waveM = wave >> 2;             // 0..1 : 32 pixels
    const int waveN = wave & 3;              // 0..3 : 32 couts
    const bool hiH  = lane >= 16;
    const int  nsel = lane & 15;
    const int  klo  = hiH ? 8  : 0;
    const int  khi  = hiH ? 24 : 16;

    // staging thread mapping
    const int pixA = threadIdx.x >> 2;       // 0..63
    const int subA = threadIdx.x & 3;        // 16-ci chunk
    const int coB  = threadIdx.x >> 1;       // 0..127
    const int subB = threadIdx.x & 1;        // 32-ci chunk

    const unsigned adst = (unsigned)(size_t)&Asmem[pixA * 72 + subA * 16];
    const unsigned bdst = (unsigned)(size_t)&Bsmem[coB * 72 + subB * 32];

    v8f c[2][2] = {};

    for (int dy = 0; dy < 3; ++dy) {
        const int  hh    = h + dy - 1;
        const bool rowOk = (hh >= 0) & (hh < HH);
        const int  hc    = rowOk ? hh : 0;
        for (int dx = 0; dx < 3; ++dx) {
            const int  ww = pixA + dx - 1;
            const bool ok = rowOk & (ww >= 0) & (ww < WW);
            const int  wc = ww < 0 ? 0 : (ww > 63 ? 63 : ww);
            // boundary lanes pull zeros from the zero row instead of branching
            const half_t* pa = ok ? xm + ((size_t)((n * HH + hc) * WW + wc) * CIN + subA * 16)
                                  : zbuf + subA * 16;
            const half_t* pb = wht + ((size_t)((dy * 3 + dx) * COUT + co0 + coB) * CIN + subB * 32);
            __builtin_prefetch(pa, 0, 1);    // global_prefetch_b8

            for (int kc = 0; kc < 8; ++kc) {
                const int ci0 = kc * 64;
                {   // B: 64 B per thread, one addr pair, offset-folded issues
                    unsigned long long gb = (unsigned long long)(size_t)(pb + ci0);
                    asm volatile(
                        "global_load_async_to_lds_b128 %0, %1, off\n\t"
                        "global_load_async_to_lds_b128 %0, %1, off offset:16\n\t"
                        "global_load_async_to_lds_b128 %0, %1, off offset:32\n\t"
                        "global_load_async_to_lds_b128 %0, %1, off offset:48"
                        :: "v"(bdst), "v"(gb) : "memory");
                }
                {   // A: 32 B per thread (zbuf reads stay in [0,512) halves)
                    unsigned long long ga = (unsigned long long)(size_t)(pa + (ok ? ci0 : ci0 & 448));
                    asm volatile(
                        "global_load_async_to_lds_b128 %0, %1, off\n\t"
                        "global_load_async_to_lds_b128 %0, %1, off offset:16"
                        :: "v"(adst), "v"(ga) : "memory");
                }
                asm volatile("s_wait_asynccnt 0x0" ::: "memory");
                __syncthreads();

                // fragments: A[m2][kf], B[set][kf]; B reused across both m2
                AFrag a[2][2], b[2][2];
                #pragma unroll
                for (int m2 = 0; m2 < 2; ++m2) {
                    const int row = waveM * 32 + m2 * 16 + nsel;
                    #pragma unroll
                    for (int kf = 0; kf < 2; ++kf) {
                        const int kb = kf * 32;
                        #pragma unroll
                        for (int v = 0; v < 4; ++v) {
                            a[m2][kf].u[v]     = *(const unsigned int*)&Asmem[row * 72 + kb + klo + 2 * v];
                            a[m2][kf].u[4 + v] = *(const unsigned int*)&Asmem[row * 72 + kb + khi + 2 * v];
                        }
                    }
                }
                #pragma unroll
                for (int set = 0; set < 2; ++set) {
                    const int col = waveN * 32 + set * 16 + nsel;
                    #pragma unroll
                    for (int kf = 0; kf < 2; ++kf) {
                        const int kb = kf * 32;
                        #pragma unroll
                        for (int v = 0; v < 4; ++v) {
                            b[set][kf].u[v]     = *(const unsigned int*)&Bsmem[col * 72 + kb + klo + 2 * v];
                            b[set][kf].u[4 + v] = *(const unsigned int*)&Bsmem[col * 72 + kb + khi + 2 * v];
                        }
                    }
                }
                #pragma unroll
                for (int m2 = 0; m2 < 2; ++m2)
                    #pragma unroll
                    for (int set = 0; set < 2; ++set)
                        #pragma unroll
                        for (int kf = 0; kf < 2; ++kf)
                            c[m2][set] = __builtin_amdgcn_wmma_f32_16x16x32_f16(
                                false, a[m2][kf].h, false, b[set][kf].h,
                                (short)0, c[m2][set], false, false);
                __syncthreads();
            }
        }
    }

    const float ns = nstr[0];
    #pragma unroll
    for (int m2 = 0; m2 < 2; ++m2) {
        #pragma unroll
        for (int set = 0; set < 2; ++set) {
            const int cout = co0 + waveN * 32 + set * 16 + nsel;
            const float dd = dbuf[n * COUT + cout];
            const float bb = cbias[cout];
            #pragma unroll
            for (int r = 0; r < 8; ++r) {
                const int w = waveM * 32 + m2 * 16 + r + (hiH ? 8 : 0);
                const int pidx = (n * HH + h) * WW + w;
                float y = c[m2][set][r] * dd + noise[pidx] * ns + bb;
                y = (y > 0.f ? y : 0.2f * y) * LRELU_GAIN;
                y = fminf(fmaxf(y, -CLIPV), CLIPV);
                out[(size_t)pidx * COUT + cout] = y;
            }
        }
    }
}

// ---------------------------------------------------------------------------
// Fallback (small-ws): modulate-on-the-fly variant (round-2 kernel).
// ---------------------------------------------------------------------------
__global__ void __launch_bounds__(256)
modconv_fallback_kernel(const float*  __restrict__ x,
                        const half_t* __restrict__ wht,
                        const float*  __restrict__ sbuf,
                        const float*  __restrict__ dbuf,
                        const float*  __restrict__ cbias,
                        const float*  __restrict__ noise,
                        const float*  __restrict__ nstr,
                        float*        __restrict__ out) {
    __shared__ __align__(16) half_t Asmem[64 * 72];
    __shared__ __align__(16) half_t Bsmem[64 * 72];

    const int bid   = blockIdx.x;
    const int tileN = bid & 7;
    const int tileM = bid >> 3;
    const int n   = tileM >> 6;
    const int h   = tileM & 63;
    const int co0 = tileN * 64;

    const int lane  = threadIdx.x & 31;
    const int wave  = threadIdx.x >> 5;
    const int waveM = wave >> 1;
    const int waveN = wave & 1;
    const bool hiH  = lane >= 16;
    const int  nsel = lane & 15;
    const int  klo  = hiH ? 8  : 0;
    const int  khi  = hiH ? 24 : 16;
    const int  arow = waveM * 16 + nsel;

    const int idx4 = threadIdx.x >> 2;
    const int sub4 = threadIdx.x & 3;
    const unsigned bdst = (unsigned)(size_t)&Bsmem[idx4 * 72 + sub4 * 16];
    const float* sp = sbuf + n * CIN + sub4 * 16;

    v8f c[2] = {};

    for (int dy = 0; dy < 3; ++dy) {
        const int  hh    = h + dy - 1;
        const bool rowOk = (hh >= 0) & (hh < HH);
        const int  hc    = rowOk ? hh : 0;
        for (int dx = 0; dx < 3; ++dx) {
            const int  ww  = idx4 + dx - 1;
            const bool ok  = rowOk & (ww >= 0) & (ww < WW);
            const float okf = ok ? 1.f : 0.f;
            const int  wc  = ww < 0 ? 0 : (ww > 63 ? 63 : ww);
            const float*  px = x + ((size_t)((n * HH + hc) * WW + wc) * CIN + sub4 * 16);
            const half_t* pw = wht + ((size_t)((dy * 3 + dx) * COUT + co0 + idx4) * CIN + sub4 * 16);
            __builtin_prefetch(px, 0, 1);

            for (int kc = 0; kc < 8; ++kc) {
                const int ci0 = kc * 64;
                {
                    unsigned long long gsrc = (unsigned long long)(size_t)(pw + ci0);
                    asm volatile(
                        "global_load_async_to_lds_b128 %0, %1, off\n\t"
                        "global_load_async_to_lds_b128 %0, %1, off offset:16"
                        :: "v"(bdst), "v"(gsrc) : "memory");
                }
                {
                    const float* pxi = px + ci0;
                    v4f x0 = *(const v4f*)(pxi);
                    v4f x1 = *(const v4f*)(pxi + 4);
                    v4f x2 = *(const v4f*)(pxi + 8);
                    v4f x3 = *(const v4f*)(pxi + 12);
                    const float* spi = sp + ci0;
                    v4f s0 = *(const v4f*)(spi)      * okf;
                    v4f s1 = *(const v4f*)(spi + 4)  * okf;
                    v4f s2 = *(const v4f*)(spi + 8)  * okf;
                    v4f s3 = *(const v4f*)(spi + 12) * okf;
                    v8h p0, p1;
                    p0[0] = (half_t)(x0.x * s0.x); p0[1] = (half_t)(x0.y * s0.y);
                    p0[2] = (half_t)(x0.z * s0.z); p0[3] = (half_t)(x0.w * s0.w);
                    p0[4] = (half_t)(x1.x * s1.x); p0[5] = (half_t)(x1.y * s1.y);
                    p0[6] = (half_t)(x1.z * s1.z); p0[7] = (half_t)(x1.w * s1.w);
                    p1[0] = (half_t)(x2.x * s2.x); p1[1] = (half_t)(x2.y * s2.y);
                    p1[2] = (half_t)(x2.z * s2.z); p1[3] = (half_t)(x2.w * s2.w);
                    p1[4] = (half_t)(x3.x * s3.x); p1[5] = (half_t)(x3.y * s3.y);
                    p1[6] = (half_t)(x3.z * s3.z); p1[7] = (half_t)(x3.w * s3.w);
                    *(v8h*)&Asmem[idx4 * 72 + sub4 * 16]     = p0;
                    *(v8h*)&Asmem[idx4 * 72 + sub4 * 16 + 8] = p1;
                }
                asm volatile("s_wait_asynccnt 0x0" ::: "memory");
                __syncthreads();

                AFrag a[2];
                #pragma unroll
                for (int kf = 0; kf < 2; ++kf) {
                    const int kb = kf * 32;
                    #pragma unroll
                    for (int v = 0; v < 4; ++v) {
                        a[kf].u[v]     = *(const unsigned int*)&Asmem[arow * 72 + kb + klo + 2 * v];
                        a[kf].u[4 + v] = *(const unsigned int*)&Asmem[arow * 72 + kb + khi + 2 * v];
                    }
                }
                #pragma unroll
                for (int set = 0; set < 2; ++set) {
                    const int col = waveN * 32 + set * 16 + nsel;
                    #pragma unroll
                    for (int kf = 0; kf < 2; ++kf) {
                        const int kb = kf * 32;
                        AFrag b;
                        #pragma unroll
                        for (int v = 0; v < 4; ++v) {
                            b.u[v]     = *(const unsigned int*)&Bsmem[col * 72 + kb + klo + 2 * v];
                            b.u[4 + v] = *(const unsigned int*)&Bsmem[col * 72 + kb + khi + 2 * v];
                        }
                        c[set] = __builtin_amdgcn_wmma_f32_16x16x32_f16(
                            false, a[kf].h, false, b.h, (short)0, c[set], false, false);
                    }
                }
                __syncthreads();
            }
        }
    }

    const float ns = nstr[0];
    #pragma unroll
    for (int set = 0; set < 2; ++set) {
        const int cout = co0 + waveN * 32 + set * 16 + nsel;
        const float dd = dbuf[n * COUT + cout];
        const float bb = cbias[cout];
        #pragma unroll
        for (int r = 0; r < 8; ++r) {
            const int w = waveM * 16 + r + (hiH ? 8 : 0);
            const int pidx = (n * HH + h) * WW + w;
            float y = c[set][r] * dd + noise[pidx] * ns + bb;
            y = (y > 0.f ? y : 0.2f * y) * LRELU_GAIN;
            y = fminf(fmaxf(y, -CLIPV), CLIPV);
            out[(size_t)pidx * COUT + cout] = y;
        }
    }
}

// ---------------------------------------------------------------------------
extern "C" void kernel_launch(void* const* d_in, const int* in_sizes, int n_in,
                              void* d_out, int out_size, void* d_ws, size_t ws_size,
                              hipStream_t stream) {
    const float* x    = (const float*)d_in[0];
    const float* dlat = (const float*)d_in[1];
    const float* aw   = (const float*)d_in[2];
    const float* ab   = (const float*)d_in[3];
    const float* cw   = (const float*)d_in[4];
    const float* cb   = (const float*)d_in[5];
    const float* nz   = (const float*)d_in[6];
    const float* nst  = (const float*)d_in[7];
    float* out = (float*)d_out;

    float*  s_buf = (float*)d_ws;                        // 16*512
    float*  d_buf = s_buf + NN * CIN;                    // 16*512
    float*  wsq   = d_buf + NN * COUT;                   // 512*512
    half_t* wht   = (half_t*)(wsq + CIN * COUT);         // 9*512*512
    half_t* zbuf  = wht + 9 * CIN * COUT;                // 512 halves of zeros
    half_t* xm    = zbuf + 512;                          // 16*64*64*512
    const size_t need = (size_t)((char*)(xm + (size_t)NN * HH * WW * CIN) - (char*)d_ws);

    affine_kernel<<<(NN * CIN) / 256, 256, 0, stream>>>(dlat, aw, ab, s_buf);
    wsq_kernel  <<<(CIN * COUT) / 256, 256, 0, stream>>>(cw, wsq);
    demod_kernel<<<(NN * COUT) / 256, 256, 0, stream>>>(s_buf, wsq, d_buf);
    w2h_kernel  <<<(9 * CIN * COUT) / 256, 256, 0, stream>>>(cw, wht);

    if (ws_size >= need) {
        zero_kernel<<<1, 256, 0, stream>>>((unsigned int*)zbuf);
        xmod_kernel<<<(NN * HH * WW * CIN / 8) / 256, 256, 0, stream>>>(x, s_buf, xm);
        // 16 samples * 64 rows * 4 cout-tiles = 4096 blocks
        modconv_async_kernel<<<NN * HH * (COUT / 128), 256, 0, stream>>>(
            xm, wht, zbuf, d_buf, cb, nz, nst, out);
    } else {
        modconv_fallback_kernel<<<NN * HH * (COUT / 64), 256, 0, stream>>>(
            x, wht, s_buf, d_buf, cb, nz, nst, out);
    }
}
